// _BaseGspace_28303834481211
// MI455X (gfx1250) — matrix operand order repaired and checked by reading, not compiled, not run on hardware
//
#include <hip/hip_runtime.h>
#include <hip/hip_bf16.h>
#include <math.h>

// Problem constants (match reference)
#define BB 8            // batches
#define NN 8192         // grid points
#define MPAD 16         // M padded to WMMA tile
#define MAX_ITER 50
#define SLICES 8        // K-split for parallelism
#define KSLICE (NN / SLICES)   // 1024
#define TILES (NN / 16)        // 512 output column tiles

typedef __attribute__((ext_vector_type(16))) __bf16 v16bf;
typedef __attribute__((ext_vector_type(8)))  float  v8f;
typedef __attribute__((ext_vector_type(4)))  int    v4i;

// ---------------- setup kernels ----------------

// log_alpha = safe_log(alpha);  f = 0
__global__ void k_setup(const float* __restrict__ alpha,
                        float* __restrict__ la, float* __restrict__ f) {
    int idx = blockIdx.x * blockDim.x + threadIdx.x;   // 0 .. BB*NN-1
    float a = alpha[idx];
    la[idx] = (a == 0.0f) ? -INFINITY : logf(a);
    f[idx] = 0.0f;
}

// expK (bf16) = exp(kernel)   -- 64M elements, runs once; L2-resident afterwards
__global__ void k_expk(const float* __restrict__ kern, __bf16* __restrict__ expK) {
    size_t stride = (size_t)gridDim.x * blockDim.x;
    for (size_t idx = (size_t)blockIdx.x * blockDim.x + threadIdx.x;
         idx < (size_t)NN * NN; idx += stride) {
        expK[idx] = (__bf16)__expf(kern[idx]);
    }
}

// ---------------- per-iteration kernels ----------------

// c[b] = max_j (0.5 f + la), -inf -> 0   (one block per batch)
__global__ void k_rowmax(const float* __restrict__ f, const float* __restrict__ la,
                         float* __restrict__ c) {
    int b = blockIdx.x;
    __shared__ float red[256];
    float m = -INFINITY;
    for (int j = threadIdx.x; j < NN; j += 256) {
        float u = 0.5f * f[b * NN + j] + la[b * NN + j];
        m = fmaxf(m, u);
    }
    red[threadIdx.x] = m;
    __syncthreads();
    for (int off = 128; off > 0; off >>= 1) {
        if (threadIdx.x < off) red[threadIdx.x] = fmaxf(red[threadIdx.x], red[threadIdx.x + off]);
        __syncthreads();
    }
    if (threadIdx.x == 0) {
        float cm = red[0];
        c[b] = isfinite(cm) ? cm : 0.0f;
    }
}

// eu[b,j] = exp(0.5 f + la - c[b]) as bf16; rows 8..15 zero (M padding)
__global__ void k_eu(const float* __restrict__ f, const float* __restrict__ la,
                     const float* __restrict__ c, __bf16* __restrict__ eu) {
    int idx = blockIdx.x * blockDim.x + threadIdx.x;   // 0 .. MPAD*NN-1
    int row = idx / NN;
    if (row < BB) {
        int j = idx - row * NN;
        float u = 0.5f * f[row * NN + j] + la[row * NN + j];
        eu[idx] = (__bf16)__expf(u - c[row]);
    } else {
        eu[idx] = (__bf16)0.0f;
    }
}

// y_part[s][b][i] = sum_{k in slice s} eu[b,k] * expK[i,k]   via WMMA bf16
// One wave per (tile, slice). A = eu (16 x 32 per step), B = expK columns
// (symmetric => contiguous row loads), C/D = 16x16 f32 accumulator.
__global__ void __launch_bounds__(256) k_gemm(const __bf16* __restrict__ expK,
                                              const __bf16* __restrict__ eu,
                                              float* __restrict__ y_part) {
    int wave = (blockIdx.x * blockDim.x + threadIdx.x) >> 5;
    int lane = threadIdx.x & 31;
    int tile = wave / SLICES;          // 0 .. TILES-1
    int s    = wave - tile * SLICES;   // 0 .. SLICES-1
    int i0   = tile * 16;
    int kbeg = s * KSLICE;

    int half = lane >> 4;              // 0 or 1
    int ln   = lane & 15;

    // A fragment: lane holds row M=ln, K chunks [h*8, h*8+8) and [16+h*8, ...)
    const __bf16* pa = eu + (size_t)ln * NN + kbeg + half * 8;
    // B fragment: lane holds col N=ln (== row i0+ln of symmetric expK),
// K chunks [h*8, h*8+8) and [16+h*8, ...)
    const __bf16* pb = expK + (size_t)(i0 + ln) * NN + kbeg + half * 8;

    v8f acc = {};
    for (int k = 0; k < KSLICE; k += 32) {
        union { v16bf v; struct { v4i lo, hi; } p; } a, b;
        a.p.lo = *(const v4i*)(pa);        // K = base .. base+7
        a.p.hi = *(const v4i*)(pa + 16);   // K = base+16 .. base+23
        b.p.lo = *(const v4i*)(pb);        // K = kb .. kb+7
        b.p.hi = *(const v4i*)(pb + 16);   // K = kb+16 .. kb+23
        acc = __builtin_amdgcn_wmma_f32_16x16x32_bf16(
            false, a.v, false, b.v, (short)0, acc, false, false);
        pa += 32;
        pb += 32;
    }

    // C layout: lanes 0-15 hold M=0..7 in acc[0..7] (batches); lanes 16-31 hold
    // the zero-padding rows M=8..15 -> discard.
    if (lane < 16) {
        float* yp = y_part + (size_t)s * BB * NN + i0 + lane;
#pragma unroll
        for (int v = 0; v < 8; ++v) yp[(size_t)v * NN] = acc[v];
    }
}

// y = sum_s y_part;  g = -2*(log(y)+c[b]);  f = 0.5*(f+g)
__global__ void k_update(const float* __restrict__ y_part, const float* __restrict__ c,
                         float* __restrict__ f) {
    int idx = blockIdx.x * blockDim.x + threadIdx.x;   // 0 .. BB*NN-1
    int b = idx / NN;
    int i = idx - b * NN;
    float y = 0.0f;
#pragma unroll
    for (int s = 0; s < SLICES; ++s) y += y_part[(size_t)s * BB * NN + (size_t)b * NN + i];
    float g = -2.0f * (logf(y) + c[b]);
    f[idx] = 0.5f * (f[idx] + g);
}

// value[b] = sum_i (-f[b,i]) * alpha[b,i]   (one block per batch)
__global__ void k_value(const float* __restrict__ f, const float* __restrict__ alpha,
                        float* __restrict__ out) {
    int b = blockIdx.x;
    __shared__ float red[256];
    float s = 0.0f;
    for (int j = threadIdx.x; j < NN; j += 256)
        s += (-f[b * NN + j]) * alpha[b * NN + j];
    red[threadIdx.x] = s;
    __syncthreads();
    for (int off = 128; off > 0; off >>= 1) {
        if (threadIdx.x < off) red[threadIdx.x] += red[threadIdx.x + off];
        __syncthreads();
    }
    if (threadIdx.x == 0) out[b] = red[0];
}

// ---------------- launcher ----------------

extern "C" void kernel_launch(void* const* d_in, const int* in_sizes, int n_in,
                              void* d_out, int out_size, void* d_ws, size_t ws_size,
                              hipStream_t stream) {
    const float* alpha = (const float*)d_in[0];   // [8, 8192]
    const float* kern  = (const float*)d_in[1];   // [8192, 8192]
    float* out = (float*)d_out;                   // [8]

    char* ws = (char*)d_ws;
    size_t off = 0;
    __bf16* expK = (__bf16*)(ws + off); off += (size_t)NN * NN * sizeof(__bf16);   // 128 MB
    __bf16* eu   = (__bf16*)(ws + off); off += (size_t)MPAD * NN * sizeof(__bf16); // 256 KB
    float* f  = (float*)(ws + off); off += (size_t)BB * NN * sizeof(float);        // 256 KB
    float* la = (float*)(ws + off); off += (size_t)BB * NN * sizeof(float);        // 256 KB
    float* yp = (float*)(ws + off); off += (size_t)SLICES * BB * NN * sizeof(float); // 2 MB
    float* c  = (float*)(ws + off); off += 256;

    // one-time per launch: log_alpha, f=0, expK = exp(kernel) in bf16 (L2-resident after)
    k_setup<<<(BB * NN) / 256, 256, 0, stream>>>(alpha, la, f);
    k_expk<<<32768, 256, 0, stream>>>(kern, expK);

    // fixed-point loop: fixed trip count (converged iterations are no-ops within tol)
    for (int it = 0; it < MAX_ITER; ++it) {
        k_rowmax<<<BB, 256, 0, stream>>>(f, la, c);
        k_eu<<<(MPAD * NN) / 256, 256, 0, stream>>>(f, la, c, eu);
        k_gemm<<<(TILES * SLICES * 32) / 256, 256, 0, stream>>>(expK, eu, yp);
        k_update<<<(BB * NN) / 256, 256, 0, stream>>>(yp, c, f);
    }

    k_value<<<BB, 256, 0, stream>>>(f, alpha, out);
}
